// SSIMLoss_73607149519412
// MI455X (gfx1250) — matrix operand order repaired and checked by reading, not compiled, not run on hardware
//
#include <hip/hip_runtime.h>

typedef __attribute__((ext_vector_type(16))) __bf16 v16bf;
typedef __attribute__((ext_vector_type(8)))  float  v8f;

#define WMMA_BF16(a, b, c) \
    __builtin_amdgcn_wmma_f32_16x16x32_bf16(false, (a), false, (b), (short)0, (c), false, false)

// ---------------- compile-time Toeplitz constant table ----------------------
// Normalized 1D Gaussian, window 11, sigma 1.5 (matches reference numpy).
constexpr float GWc[11] = {
    0.00102838f, 0.00759880f, 0.03600077f, 0.10936049f, 0.21300553f,
    0.26601207f,
    0.21300553f, 0.10936049f, 0.03600077f, 0.00759880f, 0.00102838f
};

constexpr unsigned bf16_rne(float x) {
    unsigned u = __builtin_bit_cast(unsigned int, x);
    return (u + 0x7FFFu + ((u >> 16) & 1u)) >> 16;
}
constexpr float bf16_f32(unsigned h) {
    return __builtin_bit_cast(float, h << 16);
}
// Stage-1 B (32x16): B[K][j] = GW[K-j-3]; lane: col j = L%16, K = (e/8)*16 + (L/16)*8 + (e%8).
constexpr float wgtB(int L, int e) {
    int n = L & 15, half = L >> 4;
    int d = (((e >> 3) << 4) + half * 8 + (e & 7)) - n - 3;
    return (d >= 0 && d < 11) ? GWc[d] : 0.f;
}
// Stage-2 A (16x32): A[i][K] = GW[K-i-3]; lane: row i = L%16,
// K = (L/16)*16 + e.
constexpr float wgtA(int L, int e) {
    int n = L & 15, half = L >> 4;
    int K = half * 16 + e;
    int d = K - n - 3;
    return (d >= 0 && d < 11) ? GWc[d] : 0.f;
}
struct TblT { unsigned w[32][32]; };  // per lane: [0..7]=Bh [8..15]=Bl [16..23]=Ath [24..31]=Atl
constexpr TblT make_tbl() {
    TblT t{};
    for (int L = 0; L < 32; ++L)
        for (int p = 0; p < 8; ++p) {
            float b0 = wgtB(L, 2 * p), b1 = wgtB(L, 2 * p + 1);
            float a0 = wgtA(L, 2 * p), a1 = wgtA(L, 2 * p + 1);
            unsigned b0h = bf16_rne(b0), b1h = bf16_rne(b1);
            unsigned a0h = bf16_rne(a0), a1h = bf16_rne(a1);
            t.w[L][p]      = b0h | (b1h << 16);
            t.w[L][8 + p]  = bf16_rne(b0 - bf16_f32(b0h)) | (bf16_rne(b1 - bf16_f32(b1h)) << 16);
            t.w[L][16 + p] = a0h | (a1h << 16);
            t.w[L][24 + p] = bf16_rne(a0 - bf16_f32(a0h)) | (bf16_rne(a1 - bf16_f32(a1h)) << 16);
        }
    return t;
}
__device__ __constant__ TblT CTBL = make_tbl();

__device__ __forceinline__ v16bf load_v16bf(const unsigned* p) {
    union { uint4 q[2]; v16bf v; } u;
    u.q[0] = *reinterpret_cast<const uint4*>(p);
    u.q[1] = *reinterpret_cast<const uint4*>(p + 4);
    return u.v;
}

// ---------------- per-block helpers -----------------------------------------

// Load one 16-row x 32-col raw block of pred/gt in 16-bit-A layout order.
__device__ __forceinline__ void load_block(const float* __restrict__ pp,
                                           const float* __restrict__ gp,
                                           int blockRow0, int c0, int half, int n,
                                           bool fast, float (&pr)[16], float (&gr)[16])
{
    const int row = blockRow0 + n;
    if (fast) {
        const float* prow = pp + row * 512;
        const float* grow = gp + row * 512;
#pragma unroll
        for (int sg = 0; sg < 2; ++sg) {
            const int cb = c0 - 8 + sg * 16 + half * 8;   // 8-byte aligned
#pragma unroll
            for (int k2 = 0; k2 < 4; ++k2) {
                float2 pf = *reinterpret_cast<const float2*>(prow + cb + k2 * 2);
                float2 gf = *reinterpret_cast<const float2*>(grow + cb + k2 * 2);
                pr[sg * 8 + k2 * 2]     = pf.x;
                pr[sg * 8 + k2 * 2 + 1] = pf.y;
                gr[sg * 8 + k2 * 2]     = gf.x;
                gr[sg * 8 + k2 * 2 + 1] = gf.y;
            }
        }
    } else {
        const bool rok = (row >= 0) & (row < 512);
#pragma unroll
        for (int e = 0; e < 16; ++e) {
            const int K   = ((e >> 3) << 4) + half * 8 + (e & 7);
            const int col = c0 - 8 + K;
            const bool ok = rok & (col >= 0) & (col < 512);
            pr[e] = ok ? pp[row * 512 + col] : 0.f;
            gr[e] = ok ? gp[row * 512 + col] : 0.f;
        }
    }
}

// Horizontal 11-tap blur of one 16x32 raw block: 3 WMMAs (hi/lo split).
__device__ __forceinline__ v8f stage1(const float (&q)[16], v16bf Bh, v16bf Bl)
{
    v16bf ah, al;
#pragma unroll
    for (int e = 0; e < 16; ++e) {
        float x = q[e];
        __bf16 h = (__bf16)x;
        ah[e] = h;
        al[e] = (__bf16)(x - (float)h);
    }
    v8f c;
#pragma unroll
    for (int r = 0; r < 8; ++r) c[r] = 0.f;
    c = WMMA_BF16(ah, Bh, c);
    c = WMMA_BF16(al, Bh, c);
    c = WMMA_BF16(ah, Bl, c);
    return c;
}

// Vertical 11-tap blur: repack two C-layout H blocks into B layout, 3 WMMAs.
__device__ __forceinline__ v8f stage2(const v8f& H0, const v8f& H1,
                                      v16bf Ath, v16bf Atl, int lane)
{
    // lane L (= h*16+n) needs H[K=h*16+e][n] -> block h, row e, col n,
    // living in lane (n + 16*(e/8)), VGPR (e%8)  -- VGPR idx lane-uniform.
    const int n = lane & 15;
    v16bf bh, bl;
#pragma unroll
    for (int e = 0; e < 16; ++e) {
        int src = n + ((e >= 8) ? 16 : 0);
        float h0 = __shfl(H0[e & 7], src, 32);
        float h1 = __shfl(H1[e & 7], src, 32);
        float hv = (lane < 16) ? h0 : h1;
        __bf16 hb = (__bf16)hv;
        bh[e] = hb;
        bl[e] = (__bf16)(hv - (float)hb);
    }
    v8f o;
#pragma unroll
    for (int r = 0; r < 8; ++r) o[r] = 0.f;
    o = WMMA_BF16(Ath, bh, o);
    o = WMMA_BF16(Ath, bl, o);
    o = WMMA_BF16(Atl, bh, o);
    return o;
}

// ---------------- main kernel ------------------------------------------------
// One wave walks a column strip of 8 vertically adjacent 16x16 tiles,
// carrying the 5 stage-1 accumulators so each new tile costs 1 block load
// + 30 WMMAs (vs 2 loads + 45 standalone).
#define TPW 8

__global__ __launch_bounds__(256)
void ssim_tiles_kernel(const float* __restrict__ pred,
                       const float* __restrict__ gt,
                       float* __restrict__ partial)
{
    const int tid  = threadIdx.x;
    const int lane = tid & 31;
    const int wave = tid >> 5;

    // 48 planes * 32 tile-cols * 4 strip-groups = 6144 strips (waves).
    const int gwave = blockIdx.x * 8 + wave;
    const int pl  = gwave >> 7;
    const int rem = gwave & 127;
    const int tc  = rem >> 2;
    const int tr0 = (rem & 3) << 3;       // strip start tile-row: 0,8,16,24
    const int c0  = tc << 4;

    const float* __restrict__ pp = pred + (size_t)pl * 262144u;
    const float* __restrict__ gp = gt   + (size_t)pl * 262144u;

    const int half = lane >> 4;
    const int n    = lane & 15;
    const bool cols_ok = (tc >= 1) & (tc <= 30);

    // Constant Toeplitz operands from the compile-time table (8 x b128 loads).
    const unsigned* lw = CTBL.w[lane];
    const v16bf Bh  = load_v16bf(lw);
    const v16bf Bl  = load_v16bf(lw + 8);
    const v16bf Ath = load_v16bf(lw + 16);
    const v16bf Atl = load_v16bf(lw + 24);

    const float C1 = 0.0001f;   // 0.01^2
    const float C2 = 0.0009f;   // 0.03^2
    float lsum = 0.f;

    float pr[16], gr[16];
    v8f Hc[5];

    // Preamble: lower block of first tile (rows tr0*16-8 .. +7).
    {
        const int br0 = tr0 * 16 - 8;
        load_block(pp, gp, br0, c0, half, n, (tr0 >= 1) & cols_ok, pr, gr);
#pragma unroll
        for (int q = 0; q < 5; ++q) {
            float qv[16];
#pragma unroll
            for (int e = 0; e < 16; ++e) {
                float a = pr[e], b = gr[e];
                qv[e] = (q == 0) ? a : (q == 1) ? b : (q == 2) ? a * a
                        : (q == 3) ? b * b : a * b;
            }
            Hc[q] = stage1(qv, Bh, Bl);
        }
    }

#pragma unroll 1
    for (int k = 0; k < TPW; ++k) {
        const int tr  = tr0 + k;
        const int br0 = tr * 16 + 8;      // new (upper) block rows
        load_block(pp, gp, br0, c0, half, n, (tr <= 30) & cols_ok, pr, gr);

        v8f O[5];
#pragma unroll
        for (int q = 0; q < 5; ++q) {
            float qv[16];
#pragma unroll
            for (int e = 0; e < 16; ++e) {
                float a = pr[e], b = gr[e];
                qv[e] = (q == 0) ? a : (q == 1) ? b : (q == 2) ? a * a
                        : (q == 3) ? b * b : a * b;
            }
            v8f Hn = stage1(qv, Bh, Bl);
            O[q] = stage2(Hc[q], Hn, Ath, Atl, lane);
            Hc[q] = Hn;                   // slide window down one tile
        }

#pragma unroll
        for (int r = 0; r < 8; ++r) {
            float m1 = O[0][r], m2 = O[1][r];
            float m1s = m1 * m1, m2s = m2 * m2, m12 = m1 * m2;
            float s1  = O[2][r] - m1s;
            float s2  = O[3][r] - m2s;
            float s12 = O[4][r] - m12;
            float num = (2.f * m12 + C1) * (2.f * s12 + C2);
            float den = (m1s + m2s + C1) * (s1 + s2 + C2);
            lsum += num / den;
        }
    }

    // Wave then block reduction, one partial per block.
#pragma unroll
    for (int off = 16; off > 0; off >>= 1)
        lsum += __shfl_xor(lsum, off, 32);

    __shared__ float wsum[8];
    if (lane == 0) wsum[wave] = lsum;
    __syncthreads();
    if (tid == 0) {
        float t = 0.f;
#pragma unroll
        for (int w = 0; w < 8; ++w) t += wsum[w];
        partial[blockIdx.x] = t;
    }
}

__global__ __launch_bounds__(256)
void ssim_finalize_kernel(const float* __restrict__ partial,
                          float* __restrict__ out, int nblk)
{
    float s = 0.f;
    for (int i = threadIdx.x; i < nblk; i += 256) s += partial[i];
#pragma unroll
    for (int off = 16; off > 0; off >>= 1)
        s += __shfl_xor(s, off, 32);

    __shared__ float ws[8];
    const int lane = threadIdx.x & 31, wv = threadIdx.x >> 5;
    if (lane == 0) ws[wv] = s;
    __syncthreads();
    if (threadIdx.x == 0) {
        float t = 0.f;
#pragma unroll
        for (int w = 0; w < 8; ++w) t += ws[w];
        out[0] = 1.f - t * (1.f / 12582912.f);   // 1 - mean(ssim_map)
    }
}

extern "C" void kernel_launch(void* const* d_in, const int* in_sizes, int n_in,
                              void* d_out, int out_size, void* d_ws, size_t ws_size,
                              hipStream_t stream) {
    (void)in_sizes; (void)n_in; (void)out_size; (void)ws_size;
    const float* pred = (const float*)d_in[0];
    const float* gt   = (const float*)d_in[1];
    float* out        = (float*)d_out;
    float* partial    = (float*)d_ws;       // 768 floats = 3 KB scratch

    const int nblk = 768;                   // 6144 strips / 8 waves per block
    ssim_tiles_kernel<<<nblk, 256, 0, stream>>>(pred, gt, partial);
    ssim_finalize_kernel<<<1, 256, 0, stream>>>(partial, out, nblk);
}
